// DAWN_48069273977343
// MI455X (gfx1250) — matrix operand order repaired and checked by reading, not compiled, not run on hardware
//
#include <hip/hip_runtime.h>
#include <hip/hip_bf16.h>
#include <math.h>

// ---------------------------------------------------------------------------
// MI455X / gfx1250 implementation.
//  - All dense GEMMs: bf16 WMMA (v_wmma_f32_16x16x32_bf16), f32 accumulate,
//    16x64 C tile per wave (4 accumulators -> 4 independent WMMA chains).
//  - Codebooks pre-normalized once (L2-resident, read by all token blocks).
//  - Score/top-k/gate stage fused per token in LDS (scores never hit HBM).
//  - Flash attention: one wave per 16-query tile, 32-key steps, online softmax.
// ---------------------------------------------------------------------------

typedef __attribute__((ext_vector_type(16))) __bf16 v16bf;
typedef __attribute__((ext_vector_type(8)))  float  v8f;

#define WMMA_BF16(a, b, c) \
  __builtin_amdgcn_wmma_f32_16x16x32_bf16(false, (a), false, (b), (short)0, (c), false, false)

// ---------------------------------------------------------------------------
// LayerNorm: one 256-thread block per token row.
// ---------------------------------------------------------------------------
__global__ void ln_kernel(const float* __restrict__ x, const float* __restrict__ sc,
                          const float* __restrict__ bi, float* __restrict__ out, int D) {
  __shared__ float red[256];
  int t = blockIdx.x, tid = threadIdx.x;
  const float* xr = x + (size_t)t * D;
  float s = 0.f;
  for (int i = tid; i < D; i += 256) s += xr[i];
  red[tid] = s; __syncthreads();
  for (int st = 128; st > 0; st >>= 1) { if (tid < st) red[tid] += red[tid + st]; __syncthreads(); }
  float mean = red[0] / (float)D;
  __syncthreads();
  float v = 0.f;
  for (int i = tid; i < D; i += 256) { float d = xr[i] - mean; v += d * d; }
  red[tid] = v; __syncthreads();
  for (int st = 128; st > 0; st >>= 1) { if (tid < st) red[tid] += red[tid + st]; __syncthreads(); }
  float rstd = rsqrtf(red[0] / (float)D + 1e-6f);
  float* orow = out + (size_t)t * D;
  for (int i = tid; i < D; i += 256) orow[i] = (xr[i] - mean) * rstd * sc[i] + bi[i];
}

// ---------------------------------------------------------------------------
// Row-normalize a [N,64] codebook once: out = in / (||in|| + 1e-8).
// ---------------------------------------------------------------------------
__global__ void norm_rows_kernel(const float* __restrict__ in, float* __restrict__ out) {
  __shared__ float red[64];
  int r = blockIdx.x, tid = threadIdx.x;  // 64 threads
  float v = in[(size_t)r * 64 + tid];
  red[tid] = v * v; __syncthreads();
  for (int st = 32; st > 0; st >>= 1) { if (tid < st) red[tid] += red[tid + st]; __syncthreads(); }
  float inv = 1.0f / (sqrtf(red[0]) + 1e-8f);
  out[(size_t)r * 64 + tid] = v * inv;
}

// ---------------------------------------------------------------------------
// Small dense projections (J <= 3 outputs per token): tau heads.
// ---------------------------------------------------------------------------
__global__ void tau_kernel(const float* __restrict__ A, const float* __restrict__ W,
                           const float* __restrict__ bvec, float* __restrict__ out,
                           int D, int J) {
  __shared__ float red[256];
  int t = blockIdx.x, tid = threadIdx.x;
  const float* ar = A + (size_t)t * D;
  for (int j = 0; j < J; ++j) {
    float s = 0.f;
    for (int i = tid; i < D; i += 256) s += ar[i] * W[(size_t)i * J + j];
    red[tid] = s; __syncthreads();
    for (int st = 128; st > 0; st >>= 1) { if (tid < st) red[tid] += red[tid + st]; __syncthreads(); }
    if (tid == 0) out[(size_t)t * J + j] = red[0] + bvec[j];
    __syncthreads();
  }
}

// ---------------------------------------------------------------------------
// WMMA GEMM: C[M,N] = A[M,K] * B[K,N] (+bias[N]) (+res[M,N]).
// One wave per 16x64 C tile: A fragment loaded once per K-step and reused by
// 4 WMMAs into 4 accumulators (independent dep chains). K stepped by 32, bf16.
// Fragment layouts (wave32; A/B use contiguous k, C per CDNA5 ISA 7.12.2):
//   A 16x32: lane holds row M=lane&15; VGPR j(0..7): K=half*16+2j
//   B 32x16: lane holds col N=lane&15; VGPR j: K=half*16+2j
//   C 16x16: lane holds col N=lane&15; VGPR r: M=r+8*half
// Requires M%16==0, N%64==0, K%32==0.
// ---------------------------------------------------------------------------
__global__ void gemm_wmma_bf16(const float* __restrict__ A, const float* __restrict__ B,
                               const float* __restrict__ bias, const float* __restrict__ res,
                               float* __restrict__ C, int M, int N, int K) {
  int n0 = blockIdx.x << 6;
  int m0 = blockIdx.y << 4;
  int lane = threadIdx.x & 31;
  int col  = lane & 15;
  int half = lane >> 4;
  v8f acc[4] = {v8f{}, v8f{}, v8f{}, v8f{}};
  const float* Arow = A + (size_t)(m0 + col) * K;  // A-fragment row == lane&15
  for (int k0 = 0; k0 < K; k0 += 32) {
    if (k0 + 32 < K) __builtin_prefetch(Arow + k0 + 32, 0, 1);  // global_prefetch_b8
    v16bf a;
#pragma unroll
    for (int j = 0; j < 4; ++j) {
      int k = k0 + half * 16 + 2 * j;
      a[2 * j]     = (__bf16)Arow[k];
      a[2 * j + 1] = (__bf16)Arow[k + 1];
      a[8 + 2 * j] = (__bf16)Arow[8 + k];
      a[9 + 2 * j] = (__bf16)Arow[8 + k + 1];
    }
#pragma unroll
    for (int nt = 0; nt < 4; ++nt) {
      v16bf b;
#pragma unroll
      for (int j = 0; j < 8; ++j) {
        size_t k = (size_t)(k0 + half * 16 + 2 * j);
        b[2 * j]     = (__bf16)B[k * N + n0 + nt * 16 + col];
        b[2 * j + 1] = (__bf16)B[(k + 1) * N + n0 + nt * 16 + col];
      }
      acc[nt] = WMMA_BF16(a, b, acc[nt]);
    }
  }
#pragma unroll
  for (int nt = 0; nt < 4; ++nt) {
    int n = n0 + nt * 16 + col;
    float bs = bias ? bias[n] : 0.f;
#pragma unroll
    for (int r = 0; r < 8; ++r) {
      size_t m = (size_t)(m0 + r + 8 * half);
      float v = acc[nt][r] + bs;
      if (res) v += res[m * N + n];
      C[m * N + n] = v;
    }
  }
}

// ---------------------------------------------------------------------------
// Fused score + softmax-usage + top-K + threshold gate + bottleneck emit.
// One 256-thread block per token. Scores live in dynamic LDS (N floats).
// emb must be pre-normalized (score = h . emb_i).
// ---------------------------------------------------------------------------
__global__ void gate_topk_kernel(const float* __restrict__ hmat, int hstride,
                                 const float* __restrict__ emb,
                                 const float* __restrict__ w_enc,
                                 const float* __restrict__ tau, int taustride, int tauoff,
                                 float* __restrict__ usage, float usage_w,
                                 float* __restrict__ hbn,
                                 int N, int Ksel) {
  extern __shared__ float smem[];
  float* s_scores = smem;                       // N
  float* s_h      = smem + N;                   // 64
  float* s_red    = s_h + 64;                   // 256
  int*   s_idx    = (int*)(s_red + 256);        // 256
  float* s_topv   = (float*)(s_idx + 256);      // 128
  int*   s_topi   = (int*)(s_topv + 128);       // 128
  float* s_misc   = (float*)(s_topi + 128);     // 4

  int t = blockIdx.x, tid = threadIdx.x;
  const float* hr = hmat + (size_t)t * hstride;
  if (tid < 64) s_h[tid] = hr[tid];
  __syncthreads();

  float lmax = -3.0e38f;
  for (int i = tid; i < N; i += 256) {
    const float* e = emb + (size_t)i * 64;
    float dot = 0.f;
#pragma unroll 8
    for (int d = 0; d < 64; ++d) dot += e[d] * s_h[d];
    s_scores[i] = dot;
    lmax = fmaxf(lmax, dot);
  }
  s_red[tid] = lmax; __syncthreads();
  for (int st = 128; st > 0; st >>= 1) {
    if (tid < st) s_red[tid] = fmaxf(s_red[tid], s_red[tid + st]);
    __syncthreads();
  }
  float smax = s_red[0];
  __syncthreads();

  if (usage != nullptr) {  // softmax usage accumulation (mean over tokens)
    float lsum = 0.f;
    for (int i = tid; i < N; i += 256) lsum += __expf(s_scores[i] - smax);
    s_red[tid] = lsum; __syncthreads();
    for (int st = 128; st > 0; st >>= 1) { if (tid < st) s_red[tid] += s_red[tid + st]; __syncthreads(); }
    float inv = usage_w / s_red[0];
    __syncthreads();
    for (int i = tid; i < N; i += 256) atomicAdd(&usage[i], __expf(s_scores[i] - smax) * inv);
  }
  __syncthreads();

  for (int k = 0; k < Ksel; ++k) {  // iterative top-K extraction (descending)
    float lm = -3.0e38f; int li = 0;
    for (int i = tid; i < N; i += 256) { float v = s_scores[i]; if (v > lm) { lm = v; li = i; } }
    s_red[tid] = lm; s_idx[tid] = li; __syncthreads();
    for (int st = 128; st > 0; st >>= 1) {
      if (tid < st && s_red[tid + st] > s_red[tid]) { s_red[tid] = s_red[tid + st]; s_idx[tid] = s_idx[tid + st]; }
      __syncthreads();
    }
    if (tid == 0) { s_topv[k] = s_red[0]; s_topi[k] = s_idx[0]; s_scores[s_idx[0]] = -3.0e38f; }
    __syncthreads();
  }

  float tauv = tau[(size_t)t * taustride + tauoff];
  if (tid < Ksel) {
    float raw = s_topv[tid] - tauv;
    float g = raw > 0.f ? raw : 1e-8f * __expf(raw);
    s_topv[tid] = __expf(g) - 1.f;
  }
  __syncthreads();
  if (tid == 0) {
    float gs = 1e-8f, gm = -3.0e38f;
    for (int k = 0; k < Ksel; ++k) { gs += s_topv[k]; gm = fmaxf(gm, s_topv[k]); }
    s_misc[0] = tanhf(gm) / gs;
  }
  __syncthreads();
  if (tid < 64) {
    float scale = s_misc[0], accv = 0.f;
    for (int k = 0; k < Ksel; ++k) accv += s_topv[k] * w_enc[(size_t)s_topi[k] * 64 + tid];
    hbn[(size_t)t * 64 + tid] = accv * scale;
  }
}

// ---------------------------------------------------------------------------
// Flash attention, causal, d_head = 64, wave32. One wave per (b, h, 16-query
// tile). Keys stepped 32 at a time: S = Q K^T via 2x (K=32) bf16 WMMA per
// half-tile; P (16x32, bf16 via LDS) @ V (32x16 per N-tile) is one WMMA each.
// Row softmax stats (m,l) replicated across the 16-lane half via shfl_xor.
// ---------------------------------------------------------------------------
__global__ void attention_kernel(const float* __restrict__ Q, const float* __restrict__ K,
                                 const float* __restrict__ V, float* __restrict__ O,
                                 int S, int D) {
  const int dh = 64;
  int m0 = blockIdx.x << 4;
  int h  = blockIdx.y;
  int b  = blockIdx.z;
  size_t base = (size_t)b * S * D + (size_t)h * dh;
  int lane = threadIdx.x & 31;
  int col  = lane & 15;
  int half = lane >> 4;

  __shared__ float sP[16 * 32];

  // Q fragments (A layout), prescaled by 1/sqrt(64) = 0.125
  v16bf aQ0, aQ1;
  const float* Qrow = Q + base + (size_t)(m0 + col) * D;
#pragma unroll
  for (int j = 0; j < 4; ++j) {
    int k = half * 16 + 2 * j;
    aQ0[2 * j]     = (__bf16)(0.125f * Qrow[k]);
    aQ0[2 * j + 1] = (__bf16)(0.125f * Qrow[k + 1]);
    aQ0[8 + 2 * j] = (__bf16)(0.125f * Qrow[8 + k]);
    aQ0[9 + 2 * j] = (__bf16)(0.125f * Qrow[8 + k + 1]);
    aQ1[2 * j]     = (__bf16)(0.125f * Qrow[32 + k]);
    aQ1[2 * j + 1] = (__bf16)(0.125f * Qrow[32 + k + 1]);
    aQ1[8 + 2 * j] = (__bf16)(0.125f * Qrow[40 + k]);
    aQ1[9 + 2 * j] = (__bf16)(0.125f * Qrow[40 + k + 1]);
  }

  float mrow[8], lrow[8];
  v8f ovec[4] = {v8f{}, v8f{}, v8f{}, v8f{}};
#pragma unroll
  for (int r = 0; r < 8; ++r) { mrow[r] = -1.0e30f; lrow[r] = 0.f; }

  int qlast = m0 + 15;
  for (int t0 = 0; t0 <= qlast; t0 += 32) {
    int klo = t0 + col, khi = t0 + 16 + col;
    int kloC = klo < S ? klo : S - 1;
    int khiC = khi < S ? khi : S - 1;
    const float* Kl = K + base + (size_t)kloC * D;
    const float* Kh = K + base + (size_t)khiC * D;
    v16bf bl0, bl1, bh0, bh1;
#pragma unroll
    for (int j = 0; j < 8; ++j) {
      int k = half * 16 + 2 * j;
      bl0[2 * j] = (__bf16)Kl[k];      bl0[2 * j + 1] = (__bf16)Kl[k + 1];
      bl1[2 * j] = (__bf16)Kl[32 + k]; bl1[2 * j + 1] = (__bf16)Kl[32 + k + 1];
      bh0[2 * j] = (__bf16)Kh[k];      bh0[2 * j + 1] = (__bf16)Kh[k + 1];
      bh1[2 * j] = (__bf16)Kh[32 + k]; bh1[2 * j + 1] = (__bf16)Kh[32 + k + 1];
    }
    v8f z = {};
    v8f slo = WMMA_BF16(aQ0, bl0, z); slo = WMMA_BF16(aQ1, bl1, slo);
    v8f shi = WMMA_BF16(aQ0, bh0, z); shi = WMMA_BF16(aQ1, bh1, shi);

#pragma unroll
    for (int r = 0; r < 8; ++r) {
      int q = m0 + r + 8 * half;
      float sl = (klo <= q) ? slo[r] : -1.0e30f;
      float sh = (khi <= q) ? shi[r] : -1.0e30f;
      float mx = fmaxf(sl, sh);
      for (int msk = 1; msk < 16; msk <<= 1) mx = fmaxf(mx, __shfl_xor(mx, msk, 32));
      float mnew  = fmaxf(mrow[r], mx);
      float alpha = __expf(mrow[r] - mnew);
      mrow[r] = mnew;
      float pl = __expf(sl - mnew);
      float ph = __expf(sh - mnew);
      float ps = pl + ph;
      for (int msk = 1; msk < 16; msk <<= 1) ps += __shfl_xor(ps, msk, 32);
      lrow[r] = lrow[r] * alpha + ps;
      ovec[0][r] *= alpha; ovec[1][r] *= alpha; ovec[2][r] *= alpha; ovec[3][r] *= alpha;
      sP[(r + 8 * half) * 32 + col]      = pl;
      sP[(r + 8 * half) * 32 + 16 + col] = ph;
    }
    __syncthreads();

    v16bf aP;  // P as A-matrix 16x32
    const float* Pr = sP + (size_t)col * 32;
#pragma unroll
    for (int j = 0; j < 4; ++j) {
      int k = half * 16 + 2 * j;
      aP[2 * j]     = (__bf16)Pr[k];
      aP[2 * j + 1] = (__bf16)Pr[k + 1];
      aP[8 + 2 * j] = (__bf16)Pr[8 + k];
      aP[9 + 2 * j] = (__bf16)Pr[8 + k + 1];
    }
#pragma unroll
    for (int nt = 0; nt < 4; ++nt) {
      v16bf bv;  // V as B-matrix 32x16
#pragma unroll
      for (int j = 0; j < 8; ++j) {
        int kk = t0 + half * 16 + 2 * j;
        int k1 = kk < S ? kk : S - 1;
        int k2 = (kk + 1) < S ? (kk + 1) : S - 1;
        bv[2 * j]     = (__bf16)V[base + (size_t)k1 * D + nt * 16 + col];
        bv[2 * j + 1] = (__bf16)V[base + (size_t)k2 * D + nt * 16 + col];
      }
      ovec[nt] = WMMA_BF16(aP, bv, ovec[nt]);
    }
    __syncthreads();
  }

#pragma unroll
  for (int r = 0; r < 8; ++r) {
    float inv = 1.0f / lrow[r];
    size_t off = base + (size_t)(m0 + r + 8 * half) * D + col;
    O[off]      = ovec[0][r] * inv;
    O[off + 16] = ovec[1][r] * inv;
    O[off + 32] = ovec[2][r] * inv;
    O[off + 48] = ovec[3][r] * inv;
  }
}

// ---------------------------------------------------------------------------
// Aux losses from usage accumulators.
// ---------------------------------------------------------------------------
__global__ void aux_kernel(const float* __restrict__ uqk, const float* __restrict__ uv,
                           const float* __restrict__ ukn, float* __restrict__ out) {
  __shared__ float red[256];
  int tid = threadIdx.x;
  float acc = 0.f;
  for (int i = tid; i < 8192; i += 256)  { float d = uqk[i] - 1.0f / 8192.0f;  acc += d * d * 8192.0f * 3.0f; }
  for (int i = tid; i < 8192; i += 256)  { float d = uv[i]  - 1.0f / 8192.0f;  acc += d * d * 8192.0f; }
  for (int i = tid; i < 16384; i += 256) { float d = ukn[i] - 1.0f / 16384.0f; acc += d * d * 16384.0f; }
  red[tid] = acc; __syncthreads();
  for (int st = 128; st > 0; st >>= 1) { if (tid < st) red[tid] += red[tid + st]; __syncthreads(); }
  if (tid == 0) out[0] = red[0];
}

// ---------------------------------------------------------------------------
extern "C" void kernel_launch(void* const* d_in, const int* in_sizes, int n_in,
                              void* d_out, int out_size, void* d_ws, size_t ws_size,
                              hipStream_t stream) {
  (void)in_sizes; (void)n_in; (void)out_size; (void)ws_size;
  const int Bc = 4, Sc = 1024, Dc = 1024, T = Bc * Sc;

  const float* x           = (const float*)d_in[0];
  const float* qk_emb      = (const float*)d_in[1];
  const float* v_emb       = (const float*)d_in[2];
  const float* know_emb    = (const float*)d_in[3];
  const float* qk_w_enc    = (const float*)d_in[4];
  const float* v_w_enc     = (const float*)d_in[5];
  const float* know_w_enc  = (const float*)d_in[6];
  const float* qk_w_dec    = (const float*)d_in[7];
  const float* v_w_dec     = (const float*)d_in[8];
  const float* know_w_dec  = (const float*)d_in[9];
  const float* proj_attn_k = (const float*)d_in[10];
  const float* proj_attn_b = (const float*)d_in[11];
  const float* tau_attn_k  = (const float*)d_in[12];
  const float* tau_attn_b  = (const float*)d_in[13];
  const float* proj_know_k = (const float*)d_in[14];
  const float* proj_know_b = (const float*)d_in[15];
  const float* tau_know_k  = (const float*)d_in[16];
  const float* tau_know_b  = (const float*)d_in[17];
  const float* expand_O    = (const float*)d_in[18];
  const float* n1s = (const float*)d_in[19];
  const float* n1b = (const float*)d_in[20];
  const float* n2s = (const float*)d_in[21];
  const float* n2b = (const float*)d_in[22];

  float* ws = (float*)d_ws;
  size_t off = 0;
  auto alloc = [&](size_t n) { float* p = ws + off; off += n; return p; };
  const size_t TD = (size_t)T * Dc;
  float* normed  = alloc(TD);             // reused as normed2
  float* h_all   = alloc((size_t)T * 192);
  float* tau_all = alloc((size_t)T * 3);
  float* hbnQ    = alloc((size_t)T * 64);
  float* hbnK    = alloc((size_t)T * 64);
  float* hbnV    = alloc((size_t)T * 64);
  float* Qb      = alloc(TD);
  float* Kb      = alloc(TD);
  float* Vb      = alloc(TD);
  float* attn    = alloc(TD);
  float* xmid    = alloc(TD);
  float* hk      = alloc((size_t)T * 64);
  float* tauk    = alloc((size_t)T);
  float* hbnKN   = alloc((size_t)T * 64);
  float* qk_n    = alloc((size_t)8192 * 64);   // normalized codebooks
  float* v_n     = alloc((size_t)8192 * 64);
  float* kn_n    = alloc((size_t)16384 * 64);
  float* uqk     = alloc(8192);
  float* uv      = alloc(8192);
  float* ukn     = alloc(16384);

  hipMemsetAsync(uqk, 0, (8192 + 8192 + 16384) * sizeof(float), stream);  // uqk|uv|ukn contiguous

  const float uw = 1.0f / (float)T;
  const size_t gate_sh  = ((size_t)8192  + 64 + 256 + 256 + 128 + 128 + 8) * sizeof(float);
  const size_t gate_shK = ((size_t)16384 + 64 + 256 + 256 + 128 + 128 + 8) * sizeof(float);

  // ---- normalize codebooks once (stay hot in 192MB L2 across all gates) ----
  norm_rows_kernel<<<8192, 64, 0, stream>>>(qk_emb, qk_n);
  norm_rows_kernel<<<8192, 64, 0, stream>>>(v_emb, v_n);
  norm_rows_kernel<<<16384, 64, 0, stream>>>(know_emb, kn_n);

  // ---- attention circuit ----
  ln_kernel<<<T, 256, 0, stream>>>(x, n1s, n1b, normed, Dc);
  gemm_wmma_bf16<<<dim3(192 / 64, T / 16), 32, 0, stream>>>(normed, proj_attn_k, proj_attn_b,
                                                            nullptr, h_all, T, 192, Dc);
  tau_kernel<<<T, 256, 0, stream>>>(normed, tau_attn_k, tau_attn_b, tau_all, Dc, 3);

  gate_topk_kernel<<<T, 256, gate_sh, stream>>>(h_all + 0,   192, qk_n, qk_w_enc, tau_all, 3, 0,
                                                uqk, uw, hbnQ, 8192, 64);
  gate_topk_kernel<<<T, 256, gate_sh, stream>>>(h_all + 64,  192, qk_n, qk_w_enc, tau_all, 3, 1,
                                                nullptr, 0.f, hbnK, 8192, 64);
  gate_topk_kernel<<<T, 256, gate_sh, stream>>>(h_all + 128, 192, v_n, v_w_enc, tau_all, 3, 2,
                                                uv, uw, hbnV, 8192, 64);

  gemm_wmma_bf16<<<dim3(Dc / 64, T / 16), 32, 0, stream>>>(hbnQ, qk_w_dec, nullptr, nullptr, Qb, T, Dc, 64);
  gemm_wmma_bf16<<<dim3(Dc / 64, T / 16), 32, 0, stream>>>(hbnK, qk_w_dec, nullptr, nullptr, Kb, T, Dc, 64);
  gemm_wmma_bf16<<<dim3(Dc / 64, T / 16), 32, 0, stream>>>(hbnV, v_w_dec,  nullptr, nullptr, Vb, T, Dc, 64);

  attention_kernel<<<dim3(Sc / 16, 16, Bc), 32, 0, stream>>>(Qb, Kb, Vb, attn, Sc, Dc);

  gemm_wmma_bf16<<<dim3(Dc / 64, T / 16), 32, 0, stream>>>(attn, expand_O, nullptr, x, xmid, T, Dc, Dc);

  // ---- knowledge circuit ----
  ln_kernel<<<T, 256, 0, stream>>>(xmid, n2s, n2b, normed, Dc);
  gemm_wmma_bf16<<<dim3(64 / 64, T / 16), 32, 0, stream>>>(normed, proj_know_k, proj_know_b,
                                                           nullptr, hk, T, 64, Dc);
  tau_kernel<<<T, 256, 0, stream>>>(normed, tau_know_k, tau_know_b, tauk, Dc, 1);
  gate_topk_kernel<<<T, 256, gate_shK, stream>>>(hk, 64, kn_n, know_w_enc, tauk, 1, 0,
                                                 ukn, uw, hbnKN, 16384, 128);

  float* xout = (float*)d_out;
  gemm_wmma_bf16<<<dim3(Dc / 64, T / 16), 32, 0, stream>>>(hbnKN, know_w_dec, nullptr, xmid,
                                                           xout, T, Dc, 64);
  aux_kernel<<<1, 256, 0, stream>>>(uqk, uv, ukn, xout + TD);
}
